// LinearAttention_15410342658157
// MI455X (gfx1250) — matrix operand order repaired and checked by reading, not compiled, not run on hardware
//
#include <hip/hip_runtime.h>

// LinearAttention on gfx1250 (MI455X), bf16 WMMA pipeline, fp32 accumulate.
// B=4, S=8192, D=1024, H=16, DK=64.

#define BB   4
#define SS   8192
#define DDIM 1024
#define HH   16
#define DK   64
#define MTOT (BB * SS)          // 32768 rows
#define EPSF 1e-6f

typedef __bf16 bf16_t;
typedef __attribute__((ext_vector_type(16))) __bf16 v16bf;
typedef __attribute__((ext_vector_type(8)))  __bf16 v8bf;
typedef __attribute__((ext_vector_type(8)))  float  v8f;

union FragU { v16bf f; v8bf h[2]; };

// ---------------------------------------------------------------------------
// WMMA fragment loaders (wave32).
// A matrix 16x32 bf16 (M x K), row-major source, ld in elements:
//   lane l=lane&15 holds row M=l; hi=lane>>4 selects K-halves:
//   elements 0..7  -> K = hi*8 + 0..7
//   elements 8..15 -> K = 16 + hi*8 + 0..7
__device__ __forceinline__ v16bf load_frag_a(const bf16_t* __restrict__ base, int ld) {
  const int lane = threadIdx.x & 31;
  const int l = lane & 15, hi = lane >> 4;
  const bf16_t* p = base + (size_t)l * ld + hi * 8;
  FragU u;
  u.h[0] = *(const v8bf*)(p);
  u.h[1] = *(const v8bf*)(p + 16);
  return u.f;
}

// B matrix 32x16 bf16 (K x N), sourced from B^T stored row-major [N,K]:
//   lane l holds column N=l; elements 0..7 -> K = hi*8 + 0..7, elements 8..15 -> K = 16 + hi*8 + 0..7.
// Works for both global and LDS (generic) pointers.
__device__ __forceinline__ v16bf load_frag_b(const bf16_t* baseT, int ld) {
  const int lane = threadIdx.x & 31;
  const int n = lane & 15, hi = lane >> 4;
  const bf16_t* p = baseT + (size_t)n * ld + hi * 8;
  FragU u;
  u.h[0] = *(const v8bf*)(p);
  u.h[1] = *(const v8bf*)(p + 16);
  return u.f;
}

__device__ __forceinline__ v8f wmma_bf16(v16bf a, v16bf b, v8f c) {
  return __builtin_amdgcn_wmma_f32_16x16x32_bf16(false, a, false, b, (short)0, c,
                                                 false, false);
}

// Async copy 16B global -> LDS (gfx1250, ASYNCcnt-tracked).
__device__ __forceinline__ void async_g2l_b128(unsigned lds_off, const void* gaddr) {
  asm volatile("global_load_async_to_lds_b128 %0, %1, off"
               :: "v"(lds_off), "v"(gaddr) : "memory");
}
__device__ __forceinline__ void wait_asynccnt0() {
  asm volatile("s_wait_asynccnt 0x0" ::: "memory");
}

// ---------------------------------------------------------------------------
// fp32 -> bf16 conversion (grid-stride)
__global__ void f2bf_kernel(const float* __restrict__ in, bf16_t* __restrict__ out,
                            long n) {
  long i = (long)blockIdx.x * blockDim.x + threadIdx.x;
  long stride = (long)gridDim.x * blockDim.x;
  for (; i < n; i += stride) out[i] = (bf16_t)in[i];
}

// ---------------------------------------------------------------------------
// Projection GEMM: out = relu?(X @ W^T + bias).  X[M,1024] bf16, W[N,1024] bf16.
// Block: 256 threads = 8 waves in 2(M) x 4(N); block tile 128(M) x 256(N).
// Wave tile: 64x64 -> 16 accumulators, 16 WMMAs : 16 b128 loads per K-step.
// MODE 0: Q  -> out[B,H,S,DK]            (relu + eps), direct store
// MODE 1: K  -> out[B,H,DK,S] transposed (relu + eps), LDS-coalesced store
// MODE 2: V  -> out[B,H,DK,S] transposed (no act),     LDS-coalesced store
#define TPAD 136   // LDS row stride (bf16) for 128-wide tile: avoids 256B bank period

template <int MODE>
__global__ __launch_bounds__(256) void proj_kernel(const bf16_t* __restrict__ X,
                                                   const bf16_t* __restrict__ W,
                                                   const float* __restrict__ bias,
                                                   bf16_t* __restrict__ out) {
  const int wave = threadIdx.x >> 5;
  const int lane = threadIdx.x & 31;
  const int l = lane & 15, hi = lane >> 4;
  const int wm = wave & 1;        // 2 waves along M
  const int wn = wave >> 1;       // 4 waves along N
  const int row0 = blockIdx.x * 128 + wm * 64;
  const int col0 = blockIdx.y * 256 + wn * 64;

  const v8f vzero = {0.f, 0.f, 0.f, 0.f, 0.f, 0.f, 0.f, 0.f};
  v8f acc[4][4];
#pragma unroll
  for (int i = 0; i < 4; ++i)
#pragma unroll
    for (int j = 0; j < 4; ++j) acc[i][j] = vzero;

#pragma unroll 1
  for (int kb = 0; kb < DDIM; kb += 32) {
    v16bf a[4], b[4];
#pragma unroll
    for (int i = 0; i < 4; ++i)
      a[i] = load_frag_a(X + (size_t)(row0 + 16 * i) * DDIM + kb, DDIM);
#pragma unroll
    for (int j = 0; j < 4; ++j)
      b[j] = load_frag_b(W + (size_t)(col0 + 16 * j) * DDIM + kb, DDIM);
#pragma unroll
    for (int i = 0; i < 4; ++i)
#pragma unroll
      for (int j = 0; j < 4; ++j) acc[i][j] = wmma_bf16(a[i], b[j], acc[i][j]);
  }

  if constexpr (MODE == 0) {
    // Q: direct store [B,H,S,DK]; lanes 0-15 cover consecutive d -> 32B runs.
#pragma unroll
    for (int i = 0; i < 4; ++i)
#pragma unroll
      for (int j = 0; j < 4; ++j) {
        const int n = col0 + 16 * j + l;
        const float bn = bias[n];
        const int h = n >> 6, d = n & (DK - 1);
#pragma unroll
        for (int r = 0; r < 8; ++r) {
          const int m = row0 + 16 * i + r + 8 * hi;
          float v = fmaxf(acc[i][j][r] + bn, 0.f) + EPSF;
          const int b = m >> 13, s = m & (SS - 1);
          out[(((size_t)b * HH + h) * SS + s) * DK + d] = (bf16_t)v;
        }
      }
  } else {
    // K/V: transpose through LDS, then fully-coalesced 16B vector stores.
    __shared__ bf16_t tbuf[256 * TPAD];   // [n_block 0..255][m_block 0..127]
#pragma unroll
    for (int i = 0; i < 4; ++i)
#pragma unroll
      for (int j = 0; j < 4; ++j) {
        const int nb = wn * 64 + 16 * j + l;          // n within block tile
        const float bn = bias[blockIdx.y * 256 + nb];
#pragma unroll
        for (int r = 0; r < 8; ++r) {
          const int mb = wm * 64 + 16 * i + r + 8 * hi; // m within block tile
          float v = acc[i][j][r] + bn;
          if (MODE == 1) v = fmaxf(v, 0.f) + EPSF;
          tbuf[nb * TPAD + mb] = (bf16_t)v;
        }
      }
    __syncthreads();
    // Each thread owns one n-row: 128 contiguous s-values -> 16 x v8bf stores.
    const int t = threadIdx.x;
    const int n = blockIdx.y * 256 + t;
    const int h = n >> 6, d = n & (DK - 1);
    const int m0 = blockIdx.x * 128;
    const int b = m0 >> 13, sbase = m0 & (SS - 1);
    bf16_t* dst = out + (((size_t)b * HH + h) * DK + d) * SS + sbase;
    const bf16_t* src = tbuf + (size_t)t * TPAD;
#pragma unroll
    for (int c = 0; c < 128; c += 8)
      *(v8bf*)(dst + c) = *(const v8bf*)(src + c);
  }
}

// ---------------------------------------------------------------------------
// k_sum[b,h,d] = sum_s K[b,h,s,d], from KhT [B,H,DK,S]
__global__ __launch_bounds__(64) void ksum_kernel(const bf16_t* __restrict__ KhT,
                                                  float* __restrict__ ksum) {
  const int bh = blockIdx.x;
  const int d = threadIdx.x;
  const bf16_t* p = KhT + ((size_t)bh * DK + d) * SS;
  float s = 0.f;
  for (int i = 0; i < SS; i += 8) {
    v8bf v = *(const v8bf*)(p + i);
#pragma unroll
    for (int j = 0; j < 8; ++j) s += (float)v[j];
  }
  ksum[bh * DK + d] = s;
}

// ---------------------------------------------------------------------------
// kvT[b,h,e,d] = sum_s V[s,e] * K[s,d]  ==  gemm(A=VhT[e,s], B^T=KhT[d,s])
// One block per (b,h): 8 waves = (2 e-halves) x (4 K-splits), LDS f32 reduce.
__global__ __launch_bounds__(256) void kv_kernel(const bf16_t* __restrict__ VhT,
                                                 const bf16_t* __restrict__ KhT,
                                                 bf16_t* __restrict__ kvT) {
  __shared__ float lds[DK * DK];
  const int bh = blockIdx.x;
  for (int i = threadIdx.x; i < DK * DK; i += 256) lds[i] = 0.f;
  __syncthreads();

  const int wave = threadIdx.x >> 5;
  const int lane = threadIdx.x & 31;
  const int l = lane & 15, hi = lane >> 4;
  const int e0 = (wave & 1) * 32;
  const int s0 = (wave >> 1) * (SS / 4);

  const bf16_t* A  = VhT + (size_t)bh * DK * SS;
  const bf16_t* Bt = KhT + (size_t)bh * DK * SS;

  const v8f vzero = {0.f, 0.f, 0.f, 0.f, 0.f, 0.f, 0.f, 0.f};
  v8f acc[2][4];
  for (int i = 0; i < 2; ++i)
    for (int j = 0; j < 4; ++j) acc[i][j] = vzero;

#pragma unroll 1
  for (int s = s0; s < s0 + SS / 4; s += 32) {
    v16bf a0 = load_frag_a(A + (size_t)(e0 + 0)  * SS + s, SS);
    v16bf a1 = load_frag_a(A + (size_t)(e0 + 16) * SS + s, SS);
#pragma unroll
    for (int j = 0; j < 4; ++j) {
      v16bf b = load_frag_b(Bt + (size_t)(16 * j) * SS + s, SS);
      acc[0][j] = wmma_bf16(a0, b, acc[0][j]);
      acc[1][j] = wmma_bf16(a1, b, acc[1][j]);
    }
  }

#pragma unroll
  for (int i = 0; i < 2; ++i)
#pragma unroll
    for (int j = 0; j < 4; ++j) {
      const int d = 16 * j + l;
#pragma unroll
      for (int r = 0; r < 8; ++r) {
        const int e = e0 + 16 * i + r + 8 * hi;
        atomicAdd(&lds[e * DK + d], acc[i][j][r]);   // ds_add_f32
      }
    }
  __syncthreads();
  for (int i = threadIdx.x; i < DK * DK; i += 256)
    kvT[(size_t)bh * DK * DK + i] = (bf16_t)lds[i];
}

// ---------------------------------------------------------------------------
// qkv[s,e] = (sum_d Q[s,d]*kvT[e,d]) / (sum_d Q[s,d]*ksum[d] + eps)
// kvT tile (8KB) is staged into LDS with GLOBAL_LOAD_ASYNC_TO_LDS_B128 and
// consumed as ds_load-based WMMA B-fragments by all 8 waves.
// Output interleaved bf16 attn[B,S,D] for the final projection.
__global__ __launch_bounds__(256) void qkv_kernel(const bf16_t* __restrict__ Qh,
                                                  const bf16_t* __restrict__ kvT,
                                                  const float* __restrict__ ksum,
                                                  bf16_t* __restrict__ attn) {
  __shared__ __align__(16) bf16_t kvs[DK * DK];   // 8KB staged kv tile
  __shared__ float denom[256];
  __shared__ float ks[DK];
  const int bh = blockIdx.y;
  const int s0 = blockIdx.x * 256;
  const int t = threadIdx.x;

  // Kick off async staging of the kv tile: 2 x 16B per thread (ASYNCcnt).
  {
    const char* g = (const char*)(kvT + (size_t)bh * DK * DK);
    const unsigned lbase = (unsigned)(uintptr_t)(&kvs[0]);   // LDS aperture: low 32b = offset
    async_g2l_b128(lbase + t * 16,        g + t * 16);
    async_g2l_b128(lbase + t * 16 + 4096, g + t * 16 + 4096);
  }

  if (t < DK) ks[t] = ksum[bh * DK + t];
  __syncthreads();
  {
    const bf16_t* qp = Qh + ((size_t)bh * SS + s0 + t) * DK;
    float dot = 0.f;
#pragma unroll 8
    for (int d = 0; d < DK; ++d) dot += (float)qp[d] * ks[d];
    denom[t] = dot + EPSF;
  }
  wait_asynccnt0();     // this wave's async LDS writes complete
  __syncthreads();      // all waves' writes visible block-wide

  const int wave = t >> 5;
  const int lane = t & 31;
  const int l = lane & 15, hi = lane >> 4;
  const int r0 = wave * 32;

  const bf16_t* Abase = Qh + ((size_t)bh * SS + s0 + r0) * DK;
  const bf16_t* Bt    = &kvs[0];   // LDS-resident B^T [e,d] -> ds_load frags

  const v8f vzero = {0.f, 0.f, 0.f, 0.f, 0.f, 0.f, 0.f, 0.f};
  v8f acc[2][4];
  for (int i = 0; i < 2; ++i)
    for (int j = 0; j < 4; ++j) acc[i][j] = vzero;

#pragma unroll
  for (int kb = 0; kb < DK; kb += 32) {
    v16bf a0 = load_frag_a(Abase + kb, DK);
    v16bf a1 = load_frag_a(Abase + (size_t)16 * DK + kb, DK);
#pragma unroll
    for (int j = 0; j < 4; ++j) {
      v16bf b = load_frag_b(Bt + (size_t)(16 * j) * DK + kb, DK);
      acc[0][j] = wmma_bf16(a0, b, acc[0][j]);
      acc[1][j] = wmma_bf16(a1, b, acc[1][j]);
    }
  }

  const int b = bh >> 4, h = bh & (HH - 1);
#pragma unroll
  for (int i = 0; i < 2; ++i)
#pragma unroll
    for (int j = 0; j < 4; ++j) {
      const int e = 16 * j + l;
#pragma unroll
      for (int r = 0; r < 8; ++r) {
        const int sl = r0 + 16 * i + r + 8 * hi;
        const float v = acc[i][j][r] / denom[sl];
        attn[((size_t)b * SS + (s0 + sl)) * DDIM + h * DK + e] = (bf16_t)v;
      }
    }
}

// ---------------------------------------------------------------------------
// Final projection: out = attn @ Wo^T + bo, fp32 output [B,S,D]
// Same 64x64-per-wave tiling as proj_kernel.
__global__ __launch_bounds__(256) void outproj_kernel(const bf16_t* __restrict__ X,
                                                      const bf16_t* __restrict__ W,
                                                      const float* __restrict__ bias,
                                                      float* __restrict__ out) {
  const int wave = threadIdx.x >> 5;
  const int lane = threadIdx.x & 31;
  const int l = lane & 15, hi = lane >> 4;
  const int wm = wave & 1;
  const int wn = wave >> 1;
  const int row0 = blockIdx.x * 128 + wm * 64;
  const int col0 = blockIdx.y * 256 + wn * 64;

  const v8f vzero = {0.f, 0.f, 0.f, 0.f, 0.f, 0.f, 0.f, 0.f};
  v8f acc[4][4];
#pragma unroll
  for (int i = 0; i < 4; ++i)
#pragma unroll
    for (int j = 0; j < 4; ++j) acc[i][j] = vzero;

#pragma unroll 1
  for (int kb = 0; kb < DDIM; kb += 32) {
    v16bf a[4], b[4];
#pragma unroll
    for (int i = 0; i < 4; ++i)
      a[i] = load_frag_a(X + (size_t)(row0 + 16 * i) * DDIM + kb, DDIM);
#pragma unroll
    for (int j = 0; j < 4; ++j)
      b[j] = load_frag_b(W + (size_t)(col0 + 16 * j) * DDIM + kb, DDIM);
#pragma unroll
    for (int i = 0; i < 4; ++i)
#pragma unroll
      for (int j = 0; j < 4; ++j) acc[i][j] = wmma_bf16(a[i], b[j], acc[i][j]);
  }

#pragma unroll
  for (int i = 0; i < 4; ++i)
#pragma unroll
    for (int j = 0; j < 4; ++j) {
      const int n = col0 + 16 * j + l;
      const float bn = bias[n];
#pragma unroll
      for (int r = 0; r < 8; ++r) {
        const int m = row0 + 16 * i + r + 8 * hi;
        out[(size_t)m * DDIM + n] = acc[i][j][r] + bn;
      }
    }
}

// ---------------------------------------------------------------------------
extern "C" void kernel_launch(void* const* d_in, const int* in_sizes, int n_in,
                              void* d_out, int out_size, void* d_ws, size_t ws_size,
                              hipStream_t stream) {
  const float* x  = (const float*)d_in[0];
  const float* Wq = (const float*)d_in[1];
  const float* bq = (const float*)d_in[2];
  const float* Wk = (const float*)d_in[3];
  const float* bk = (const float*)d_in[4];
  const float* Wv = (const float*)d_in[5];
  const float* bv = (const float*)d_in[6];
  const float* Wo = (const float*)d_in[7];
  const float* bo = (const float*)d_in[8];
  float* out = (float*)d_out;

  // Workspace partition (aligned to 256B).
  size_t off = 0;
  auto carve = [&](size_t bytes) -> void* {
    void* p = (char*)d_ws + off;
    off += (bytes + 255) & ~(size_t)255;
    return p;
  };
  const size_t actN = (size_t)MTOT * DDIM;       // 33.5M elements
  const size_t wN   = (size_t)DDIM * DDIM;       // 1M elements
  bf16_t* xbf  = (bf16_t*)carve(actN * sizeof(bf16_t));
  bf16_t* wqb  = (bf16_t*)carve(wN * sizeof(bf16_t));
  bf16_t* wkb  = (bf16_t*)carve(wN * sizeof(bf16_t));
  bf16_t* wvb  = (bf16_t*)carve(wN * sizeof(bf16_t));
  bf16_t* wob  = (bf16_t*)carve(wN * sizeof(bf16_t));
  bf16_t* Qh   = (bf16_t*)carve(actN * sizeof(bf16_t));   // [B,H,S,DK]
  bf16_t* KhT  = (bf16_t*)carve(actN * sizeof(bf16_t));   // [B,H,DK,S]
  bf16_t* VhT  = (bf16_t*)carve(actN * sizeof(bf16_t));   // [B,H,DK,S]
  bf16_t* kvT  = (bf16_t*)carve((size_t)BB * HH * DK * DK * sizeof(bf16_t));
  float*  ksum = (float*)carve((size_t)BB * HH * DK * sizeof(float));
  bf16_t* attn = xbf;  // reuse: xbf dead after the three projections

  // 1) fp32 -> bf16
  f2bf_kernel<<<2048, 256, 0, stream>>>(x,  xbf, (long)actN);
  f2bf_kernel<<<512,  256, 0, stream>>>(Wq, wqb, (long)wN);
  f2bf_kernel<<<512,  256, 0, stream>>>(Wk, wkb, (long)wN);
  f2bf_kernel<<<512,  256, 0, stream>>>(Wv, wvb, (long)wN);
  f2bf_kernel<<<512,  256, 0, stream>>>(Wo, wob, (long)wN);

  // 2) Q/K/V projections (WMMA)
  dim3 gproj(MTOT / 128, DDIM / 256);
  proj_kernel<0><<<gproj, 256, 0, stream>>>(xbf, wqb, bq, Qh);
  proj_kernel<1><<<gproj, 256, 0, stream>>>(xbf, wkb, bk, KhT);
  proj_kernel<2><<<gproj, 256, 0, stream>>>(xbf, wvb, bv, VhT);

  // 3) k_sum and kv state
  ksum_kernel<<<BB * HH, 64, 0, stream>>>(KhT, ksum);
  kv_kernel<<<BB * HH, 256, 0, stream>>>(VhT, KhT, kvT);

  // 4) qkv + normalization -> interleaved bf16 [B,S,D]
  dim3 gq(SS / 256, BB * HH);
  qkv_kernel<<<gq, 256, 0, stream>>>(Qh, kvT, ksum, attn);

  // 5) output projection -> fp32
  outproj_kernel<<<gproj, 256, 0, stream>>>(attn, wob, bo, out);
}
